// SemiPermeableAttention_41351945126015
// MI455X (gfx1250) — matrix operand order repaired and checked by reading, not compiled, not run on hardware
//
#include <hip/hip_runtime.h>
#include <hip/hip_bf16.h>

// =====================================================================
// Causal MHA for MI455X (gfx1250), bf16 WMMA end-to-end, with
// async global->LDS tile staging (ASYNCcnt path) + prefetch.
// B=4, S=2048, D=1024, H=16, dh=64.
// Workspace requirement: ~109 MB (54,525,952 bf16 elements).
// =====================================================================

#define BATCH 4
#define SEQ   2048
#define DMODEL 1024
#define NHEAD 16
#define DHEAD 64
#define MROWS (BATCH * SEQ)   // 8192

typedef __attribute__((ext_vector_type(16))) __bf16          v16bf;
typedef __attribute__((ext_vector_type(8)))  float           v8f;
typedef __attribute__((ext_vector_type(16))) unsigned short  v16u;
typedef __attribute__((ext_vector_type(8)))  unsigned short  v8u;

// ---- float -> bf16 (round to nearest even), raw-bits representation ----
__device__ __forceinline__ unsigned short f2bf(float f) {
  unsigned int u = __builtin_bit_cast(unsigned int, f);
  unsigned int r = u + 0x7fffu + ((u >> 16) & 1u);
  return (unsigned short)(r >> 16);
}

// Assemble a 16-element bf16 fragment from two 16-byte LDS loads.
__device__ __forceinline__ v16bf mk_frag(const unsigned short* p0,
                                         const unsigned short* p1) {
  v8u a = *(const v8u*)p0;
  v8u b = *(const v8u*)p1;
  v16u t = __builtin_shufflevector(a, b, 0, 1, 2, 3, 4, 5, 6, 7,
                                         8, 9, 10, 11, 12, 13, 14, 15);
  return __builtin_bit_cast(v16bf, t);
}

__device__ __forceinline__ v8f wmma_bf16(v16bf a, v16bf b, v8f c) {
  // v_wmma_f32_16x16x32_bf16, fp32 accumulate
  return __builtin_amdgcn_wmma_f32_16x16x32_bf16(
      false, a, false, b, (short)0, c, false, false);
}

// ---- CDNA5 async global->LDS copy (bypasses VGPRs; tracked by ASYNCcnt) ---
// Generic shared pointers on amdgcn carry the LDS byte offset in the low
// 32 bits (ISA 10.2).
__device__ __forceinline__ void async_g2l_b128(void* lds_dst, const void* gsrc) {
  unsigned int lds_off = (unsigned int)(size_t)lds_dst;
  asm volatile("global_load_async_to_lds_b128 %0, %1, off"
               :
               : "v"(lds_off), "v"(gsrc)
               : "memory");
}
__device__ __forceinline__ void wait_async0() {
  asm volatile("s_wait_asynccnt 0" ::: "memory");
}

// =====================================================================
// fp32 -> bf16 conversion
// =====================================================================
__global__ __launch_bounds__(256) void cvt_f32_bf16(
    const float* __restrict__ src, unsigned short* __restrict__ dst, int n) {
  int i = blockIdx.x * blockDim.x + threadIdx.x;
  int stride = gridDim.x * blockDim.x;
  for (; i < n; i += stride) dst[i] = f2bf(src[i]);
}

// =====================================================================
// GEMM:  Y[M,N] = X[M,K] @ W[N,K]^T + bias[N]
// Block tile 128x128x32, 256 threads = 8 waves (4 in M, 2 in N),
// each wave: 32x64 tile = 2x4 WMMA accumulators.
// MODE 0: store bf16 row-major [M,N]
// MODE 1: store fp32 row-major [M,N]
// MODE 2: store bf16 transposed for V: out[((b*H+h)*64+dim)*S + s]
// =====================================================================
template <int MODE>
__global__ __launch_bounds__(256) void gemm_bf16_wmma(
    const unsigned short* __restrict__ X,   // [M,K] bf16
    const unsigned short* __restrict__ W,   // [N,K] bf16
    const float* __restrict__ bias,         // [N]
    void* __restrict__ Out,
    int M, int N, int K) {
  __shared__ __align__(16) unsigned short sA[128 * 48];  // [row][k] pad 48
  __shared__ __align__(16) unsigned short sB[128 * 48];  // [n]  [k] pad 48

  const int tid = threadIdx.x;
  const int w = tid >> 5;
  const int lane = tid & 31;
  const int sel = lane >> 4;   // half-wave select
  const int lc = lane & 15;
  const int wm = w & 3;        // 4 waves along M
  const int wn = w >> 2;       // 2 waves along N
  const int m0 = blockIdx.y * 128;
  const int n0 = blockIdx.x * 128;

  v8f acc[2][4] = {};

  for (int k0 = 0; k0 < K; k0 += 32) {
    // ---- async stage of 128x32 bf16 tiles for A and B (no VGPR round trip)
#pragma unroll
    for (int i = 0; i < 2; ++i) {
      int c = tid + i * 256;      // 0..511
      int row = c >> 2;           // 0..127
      int c8 = c & 3;             // 8-elem chunk
      async_g2l_b128(&sA[row * 48 + c8 * 8],
                     X + (size_t)(m0 + row) * K + k0 + c8 * 8);
      async_g2l_b128(&sB[row * 48 + c8 * 8],
                     W + (size_t)(n0 + row) * K + k0 + c8 * 8);
    }
    // warm L2/WGP$ for the next K-step while this tile is consumed
    if (k0 + 32 < K) {
      __builtin_prefetch(X + (size_t)(m0 + (tid >> 1)) * K + k0 + 32, 0, 0);
      __builtin_prefetch(W + (size_t)(n0 + (tid >> 1)) * K + k0 + 32, 0, 0);
    }
    wait_async0();
    __syncthreads();

    // ---- fragments + WMMA ----
    v16bf afr[2], bfr[4];
#pragma unroll
    for (int mi = 0; mi < 2; ++mi) {
      const unsigned short* p = &sA[(wm * 32 + mi * 16 + lc) * 48 + sel * 16];
      afr[mi] = mk_frag(p, p + 8);           // A: contiguous 16 K per half-wave
    }
#pragma unroll
    for (int ni = 0; ni < 4; ++ni) {
      const unsigned short* p = &sB[(wn * 64 + ni * 16 + lc) * 48 + sel * 16];
      bfr[ni] = mk_frag(p, p + 8);           // B: contiguous 16 K per half-wave
    }
#pragma unroll
    for (int mi = 0; mi < 2; ++mi)
#pragma unroll
      for (int ni = 0; ni < 4; ++ni)
        acc[mi][ni] = wmma_bf16(afr[mi], bfr[ni], acc[mi][ni]);
    __syncthreads();
  }

  // ---- epilogue: bias add + store ----
#pragma unroll
  for (int mi = 0; mi < 2; ++mi) {
#pragma unroll
    for (int ni = 0; ni < 4; ++ni) {
      int col = n0 + wn * 64 + ni * 16 + lc;
      float bv = bias[col];
      union { v8f v; float f[8]; } u;
      u.v = acc[mi][ni];
#pragma unroll
      for (int r = 0; r < 8; ++r) {
        int row = m0 + wm * 32 + mi * 16 + r + (sel ? 8 : 0);
        float val = u.f[r] + bv;
        if (MODE == 1) {
          ((float*)Out)[(size_t)row * N + col] = val;
        } else if (MODE == 0) {
          ((unsigned short*)Out)[(size_t)row * N + col] = f2bf(val);
        } else {  // MODE 2: V transposed [B,H,dh,S]
          int b = row >> 11;          // row / SEQ
          int s = row & (SEQ - 1);
          int h = col >> 6;           // col / DHEAD
          int dim = col & (DHEAD - 1);
          ((unsigned short*)Out)[(((size_t)(b * NHEAD + h) * DHEAD) + dim) * SEQ + s] =
              f2bf(val);
        }
      }
    }
  }
}

// =====================================================================
// Flash attention (causal, additive -10000 mask like the reference).
// grid = (S/128, B*H), block = 256 (8 waves). Wave w owns 16 q rows.
// KV tiles of 32 keys async-staged in LDS; online softmax in fp32.
// =====================================================================
__global__ __launch_bounds__(256) void flash_attn_wmma(
    const unsigned short* __restrict__ Q,    // [B,S,D] bf16
    const unsigned short* __restrict__ Kb,   // [B,S,D] bf16
    const unsigned short* __restrict__ Vt,   // [B,H,dh,S] bf16
    unsigned short* __restrict__ O) {        // [B,S,D] bf16
  __shared__ __align__(16) unsigned short sK[32 * 80];   // [key][dim], pad 80
  __shared__ __align__(16) unsigned short sV[64 * 48];   // [dim][key], pad 48
  __shared__ __align__(16) unsigned short sP[8][16 * 32];  // per-wave P scratch

  const int bh = blockIdx.y;
  const int b = bh >> 4;
  const int h = bh & 15;
  const int q0 = blockIdx.x * 128;
  const int tid = threadIdx.x;
  const int w = tid >> 5;
  const int lane = tid & 31;
  const int sel = lane >> 4;
  const int lc = lane & 15;
  const int qr0 = q0 + w * 16;

  // Q fragments (held in registers for whole kernel): 16 rows x 64 dh
  const unsigned short* qbase =
      Q + (size_t)(b * SEQ + qr0 + lc) * DMODEL + h * DHEAD;
  v16bf qf[2];
  qf[0] = mk_frag(qbase + sel * 16, qbase + sel * 16 + 8);            // dh 0..31
  qf[1] = mk_frag(qbase + 32 + sel * 16, qbase + 32 + sel * 16 + 8);  // dh 32..63

  float m_run[8], l_run[8];
#pragma unroll
  for (int r = 0; r < 8; ++r) { m_run[r] = -1e30f; l_run[r] = 0.f; }
  v8f oacc[4] = {};

  const float scale = 0.125f;  // 1/sqrt(64)
  const int kv_end = q0 + 128;

  for (int kv0 = 0; kv0 < kv_end; kv0 += 32) {
    // ---- async stage K tile (32 keys x 64 dims) and V tile (64 x 32) ----
    {
      int key = tid >> 3, c8 = tid & 7;
      async_g2l_b128(&sK[key * 80 + c8 * 8],
                     Kb + (size_t)(b * SEQ + kv0 + key) * DMODEL +
                         h * DHEAD + c8 * 8);
    }
    {
      int dim = tid >> 2, c4 = tid & 3;
      async_g2l_b128(&sV[dim * 48 + c4 * 8],
                     Vt + ((size_t)(b * NHEAD + h) * DHEAD + dim) * SEQ +
                         kv0 + c4 * 8);
    }
    wait_async0();
    __syncthreads();

    // ---- scores: Q(16x64) x K_tile^T(64x32) -> two 16x16 blocks ----
    v8f s0 = {}, s1 = {};
#pragma unroll
    for (int kk = 0; kk < 2; ++kk) {
      const unsigned short* p0 = &sK[lc * 80 + kk * 32 + sel * 16];
      const unsigned short* p1 = &sK[(16 + lc) * 80 + kk * 32 + sel * 16];
      s0 = wmma_bf16(qf[kk], mk_frag(p0, p0 + 8), s0);
      s1 = wmma_bf16(qf[kk], mk_frag(p1, p1 + 8), s1);
    }

    // ---- online softmax over the 32 keys ----
    union { v8f v; float f[8]; } u0, u1;
    u0.v = s0; u1.v = s1;
    float p0f[8], p1f[8], corr[8];
    int key0 = kv0 + lc, key1 = kv0 + 16 + lc;
#pragma unroll
    for (int r = 0; r < 8; ++r) {
      int q_idx = qr0 + r + (sel ? 8 : 0);
      float v0 = u0.f[r] * scale + ((key0 > q_idx) ? -10000.f : 0.f);
      float v1 = u1.f[r] * scale + ((key1 > q_idx) ? -10000.f : 0.f);
      float rm = fmaxf(v0, v1);
      rm = fmaxf(rm, __shfl_xor(rm, 1, 32));
      rm = fmaxf(rm, __shfl_xor(rm, 2, 32));
      rm = fmaxf(rm, __shfl_xor(rm, 4, 32));
      rm = fmaxf(rm, __shfl_xor(rm, 8, 32));
      float nm = fmaxf(m_run[r], rm);
      float e0 = __expf(v0 - nm);
      float e1 = __expf(v1 - nm);
      float rs = e0 + e1;
      rs += __shfl_xor(rs, 1, 32);
      rs += __shfl_xor(rs, 2, 32);
      rs += __shfl_xor(rs, 4, 32);
      rs += __shfl_xor(rs, 8, 32);
      corr[r] = __expf(m_run[r] - nm);
      l_run[r] = l_run[r] * corr[r] + rs;
      m_run[r] = nm;
      p0f[r] = e0;
      p1f[r] = e1;
    }

    // rescale running output accumulators
#pragma unroll
    for (int ni = 0; ni < 4; ++ni) {
      union { v8f v; float f[8]; } uo;
      uo.v = oacc[ni];
#pragma unroll
      for (int r = 0; r < 8; ++r) uo.f[r] *= corr[r];
      oacc[ni] = uo.v;
    }

    // ---- transpose P (C-layout) to A-layout via per-wave LDS scratch ----
#pragma unroll
    for (int r = 0; r < 8; ++r) {
      int row = r + (sel ? 8 : 0);
      sP[w][row * 32 + lc] = f2bf(p0f[r]);
      sP[w][row * 32 + 16 + lc] = f2bf(p1f[r]);
    }
    const unsigned short* pp = &sP[w][lc * 32 + sel * 16];
    v16bf pf = mk_frag(pp, pp + 8);  // P: 16 q-rows x 32 keys (A layout)

    // ---- PV: P(16x32) x V_tile(32 keys x 64 dims) ----
#pragma unroll
    for (int ni = 0; ni < 4; ++ni) {
      const unsigned short* vp = &sV[(ni * 16 + lc) * 48 + sel * 16];
      oacc[ni] = wmma_bf16(pf, mk_frag(vp, vp + 8), oacc[ni]);
    }
    __syncthreads();
  }

  // ---- normalize and store bf16 attention output [B,S,D] ----
#pragma unroll
  for (int ni = 0; ni < 4; ++ni) {
    union { v8f v; float f[8]; } uo;
    uo.v = oacc[ni];
#pragma unroll
    for (int r = 0; r < 8; ++r) {
      int qrow = qr0 + r + (sel ? 8 : 0);
      int dim = ni * 16 + lc;
      float val = uo.f[r] / l_run[r];
      O[(size_t)(b * SEQ + qrow) * DMODEL + h * DHEAD + dim] = f2bf(val);
    }
  }
}

// =====================================================================
// Host-side orchestration
// =====================================================================
extern "C" void kernel_launch(void* const* d_in, const int* in_sizes, int n_in,
                              void* d_out, int out_size, void* d_ws, size_t ws_size,
                              hipStream_t stream) {
  (void)in_sizes; (void)n_in; (void)out_size; (void)ws_size;

  const float* x_q = (const float*)d_in[0];
  const float* x_kv = (const float*)d_in[1];
  const float* Wq = (const float*)d_in[2];
  const float* bq = (const float*)d_in[3];
  const float* Wk = (const float*)d_in[4];
  const float* bk = (const float*)d_in[5];
  const float* Wv = (const float*)d_in[6];
  const float* bv = (const float*)d_in[7];
  const float* Wo = (const float*)d_in[8];
  const float* bo = (const float*)d_in[9];

  const size_t XE = (size_t)MROWS * DMODEL;   // 8,388,608
  const size_t WE = (size_t)DMODEL * DMODEL;  // 1,048,576

  unsigned short* ws = (unsigned short*)d_ws;
  size_t o = 0;
  unsigned short* xq_bf = ws + o;  o += XE;
  unsigned short* xkv_bf = ws + o; o += XE;
  unsigned short* wq_bf = ws + o;  o += WE;
  unsigned short* wk_bf = ws + o;  o += WE;
  unsigned short* wv_bf = ws + o;  o += WE;
  unsigned short* wo_bf = ws + o;  o += WE;
  unsigned short* Qb = ws + o;     o += XE;
  unsigned short* Kbf = ws + o;    o += XE;
  unsigned short* Vt = ws + o;     o += XE;
  unsigned short* Ab = ws + o;     o += XE;

  // 1) fp32 -> bf16 conversions
  cvt_f32_bf16<<<2048, 256, 0, stream>>>(x_q, xq_bf, (int)XE);
  cvt_f32_bf16<<<2048, 256, 0, stream>>>(x_kv, xkv_bf, (int)XE);
  cvt_f32_bf16<<<1024, 256, 0, stream>>>(Wq, wq_bf, (int)WE);
  cvt_f32_bf16<<<1024, 256, 0, stream>>>(Wk, wk_bf, (int)WE);
  cvt_f32_bf16<<<1024, 256, 0, stream>>>(Wv, wv_bf, (int)WE);
  cvt_f32_bf16<<<1024, 256, 0, stream>>>(Wo, wo_bf, (int)WE);

  // 2) QKV projections (V stored transposed for PV fragment locality)
  dim3 gg(DMODEL / 128, MROWS / 128);  // (8, 64)
  gemm_bf16_wmma<0><<<gg, 256, 0, stream>>>(xq_bf, wq_bf, bq, Qb,
                                            MROWS, DMODEL, DMODEL);
  gemm_bf16_wmma<0><<<gg, 256, 0, stream>>>(xkv_bf, wk_bf, bk, Kbf,
                                            MROWS, DMODEL, DMODEL);
  gemm_bf16_wmma<2><<<gg, 256, 0, stream>>>(xkv_bf, wv_bf, bv, Vt,
                                            MROWS, DMODEL, DMODEL);

  // 3) causal flash attention
  dim3 ga(SEQ / 128, BATCH * NHEAD);   // (16, 64)
  flash_attn_wmma<<<ga, 256, 0, stream>>>(Qb, Kbf, Vt, Ab);

  // 4) output projection, fp32 result
  gemm_bf16_wmma<1><<<gg, 256, 0, stream>>>(Ab, wo_bf, bo, (float*)d_out,
                                            MROWS, DMODEL, DMODEL);
}
